// TransformerDecoderLayer_64759516889112
// MI455X (gfx1250) — compile-verified, matrix operand order checked by reading, not run on hardware
//
#include <hip/hip_runtime.h>

// ---------------------------------------------------------------------------
// CDNA5 / gfx1250 transformer decoder layer.
// - All GEMMs on v_wmma_f32_16x16x32_bf16 (fp32 accumulate).
// - Fused flash attention (no S x S score materialization: saves ~800MB HBM).
// - Global->LDS staging uses GLOBAL_LOAD_ASYNC_TO_LDS_B128 + s_wait_asynccnt
//   double buffering when the toolchain exposes the builtin; otherwise falls
//   back to sync load+ds_store (still double-buffered).
// ---------------------------------------------------------------------------

typedef __bf16 bf16;
typedef bf16  v16bf __attribute__((ext_vector_type(16)));
typedef float v8f   __attribute__((ext_vector_type(8)));
typedef int   b128v __attribute__((vector_size(16)));   // int4 payload type

#define D_MODEL 768
#define D_FF    3072
#define SEQ     2048
#define BATCH   2
#define NTOK    (BATCH * SEQ)
#define NHEAD   12
#define DKH     64

#if __has_builtin(__builtin_amdgcn_global_load_async_to_lds_b128)
#define HAVE_ASYNC_LDS 1
#else
#define HAVE_ASYNC_LDS 0
#endif

// Async DMA global -> LDS (128b per lane). Falls back to sync copy.
// Builtin signature (from hipcc diagnostic): param0 = int4 addrspace(1)*,
// then lds ptr, imm offset, imm cpol.
__device__ __forceinline__ void g2l_b128(void* lds, const void* g) {
#if HAVE_ASYNC_LDS
  typedef b128v __attribute__((address_space(1)))* gp_t;
  typedef b128v __attribute__((address_space(3)))* lp_t;
  __builtin_amdgcn_global_load_async_to_lds_b128(
      (gp_t)(unsigned long long)g,
      (lp_t)(unsigned)(unsigned long long)lds,
      0, 0);
#else
  *(uint4*)lds = *(const uint4*)g;
#endif
}

template <int N> __device__ __forceinline__ void wait_async() {
#if HAVE_ASYNC_LDS
#if __has_builtin(__builtin_amdgcn_s_wait_asynccnt)
  __builtin_amdgcn_s_wait_asynccnt(N);
#else
  asm volatile("s_wait_asynccnt %0" ::"n"(N) : "memory");
#endif
#endif
}

__device__ __forceinline__ bf16 f2bf(float f) {
  union { float f; unsigned u; } a; a.f = f;
  unsigned r = a.u + 0x7FFFu + ((a.u >> 16) & 1u);   // round-to-nearest-even
  union { unsigned short s; bf16 h; } o; o.s = (unsigned short)(r >> 16);
  return o.h;
}

__device__ __forceinline__ v8f wmma_bf16(v16bf a, v16bf b, v8f c) {
  return __builtin_amdgcn_wmma_f32_16x16x32_bf16(false, a, false, b,
                                                 (short)0, c, false, false);
}

// 16x32 bf16 fragment, K-contiguous rows (CDNA5 ISA 7.12.2 16-bit A layout):
// lanes 0-15 -> idx=lane, K {0..7,16..23}; lanes 16-31 -> idx=lane-16,
// K {8..15,24..31}. Serves A, and B when W is stored [N,K] row-major.
__device__ __forceinline__ v16bf frag_k_major(const bf16* base, int ld) {
  int lane = threadIdx.x & 31;
  int idx  = lane & 15;
  int half = lane >> 4;
  const bf16* r = base + idx * ld + half * 8;
  union { uint4 q[2]; v16bf v; } t;
  t.q[0] = *(const uint4*)(r);        // K = half*8 + 0..7
  t.q[1] = *(const uint4*)(r + 16);   // K = 16 + half*8 + 0..7
  return t.v;
}

// B fragment from [K][N] row-major storage (element (k,n) = base[k*ld+n]).
__device__ __forceinline__ v16bf frag_b_colmajor(const bf16* base, int ld) {
  int lane = threadIdx.x & 31;
  int n    = lane & 15;
  int half = lane >> 4;
  v16bf f;
#pragma unroll
  for (int v2 = 0; v2 < 8; ++v2) {
    int k0 = ((v2 < 4) ? (2 * v2) : (8 + 2 * v2)) + half * 8;
    f[2 * v2]     = base[(size_t)k0 * ld + n];
    f[2 * v2 + 1] = base[(size_t)(k0 + 1) * ld + n];
  }
  return f;
}

// ---------------------------------------------------------------------------
__global__ void cvt_f32_bf16(const float* __restrict__ in,
                             bf16* __restrict__ out, int n) {
  int i = blockIdx.x * blockDim.x + threadIdx.x;
  if (i < n) out[i] = f2bf(in[i]);
}

// ---------------------------------------------------------------------------
// C[M,N] = A[M,K] @ W[N,K]^T + bias, optional ReLU, f32 and/or bf16 out.
// WG 256 threads = 8 waves; tile 128(M) x 128(N); K-step 32, double-buffered
// async staging. Each wave: 32x64 = 2x4 WMMA accumulators (8 WMMA / K-step).
// ---------------------------------------------------------------------------
__global__ __launch_bounds__(256) void gemm_bf16_wmma(
    const bf16* __restrict__ A, const bf16* __restrict__ W,
    const float* __restrict__ bias,
    float* __restrict__ Cf, bf16* __restrict__ Cb,
    int M, int N, int K, int relu) {
  __shared__ __align__(16) bf16 sA[2][128 * 32];
  __shared__ __align__(16) bf16 sB[2][128 * 32];

  const int tiles_n = N >> 7;
  const int bm = (blockIdx.x / tiles_n) << 7;
  const int bn = (blockIdx.x % tiles_n) << 7;
  const int wave = threadIdx.x >> 5;
  const int wm = (wave & 3) * 32;
  const int wn = (wave >> 2) * 64;
  const int lane = threadIdx.x & 31, half = lane >> 4, nl = lane & 15;

  // stage one 128x32 A tile + 128x32 B tile (4 x b128 per thread)
  auto stage = [&](int t, int buf) {
    const int k0 = t << 5;
#pragma unroll
    for (int i = 0; i < 2; ++i) {
      int idx = threadIdx.x * 16 + i * 8;
      int r = idx >> 5, c = idx & 31;
      g2l_b128(&sA[buf][idx], &A[(size_t)(bm + r) * K + k0 + c]);
      g2l_b128(&sB[buf][idx], &W[(size_t)(bn + r) * K + k0 + c]);
    }
  };

  v8f acc[2][4] = {};
  const int T = K >> 5;
  stage(0, 0);
  for (int t = 0; t < T; ++t) {
    const int cur = t & 1;
    if (t + 1 < T) { stage(t + 1, cur ^ 1); wait_async<4>(); }
    else           { wait_async<0>(); }
    __syncthreads();

    v16bf af[2], bfr[4];
    af[0] = frag_k_major(&sA[cur][(wm + 0) * 32], 32);
    af[1] = frag_k_major(&sA[cur][(wm + 16) * 32], 32);
#pragma unroll
    for (int j = 0; j < 4; ++j)
      bfr[j] = frag_k_major(&sB[cur][(wn + j * 16) * 32], 32);
#pragma unroll
    for (int i = 0; i < 2; ++i)
#pragma unroll
      for (int j = 0; j < 4; ++j)
        acc[i][j] = wmma_bf16(af[i], bfr[j], acc[i][j]);
    __syncthreads();
  }

  // Epilogue: C layout -> VGPR r holds M=r(+8 upper half), N=lane&15.
#pragma unroll
  for (int i = 0; i < 2; ++i)
#pragma unroll
    for (int j = 0; j < 4; ++j)
#pragma unroll
      for (int r = 0; r < 8; ++r) {
        int m = bm + wm + i * 16 + r + 8 * half;
        int n = bn + wn + j * 16 + nl;
        float v = acc[i][j][r] + (bias ? bias[n] : 0.0f);
        if (relu) v = fmaxf(v, 0.0f);
        if (Cf) Cf[(size_t)m * N + n] = v;
        if (Cb) Cb[(size_t)m * N + n] = f2bf(v);
      }
}

// ---------------------------------------------------------------------------
// Flash attention: wave = 16 q rows x d_k=64; WG = 8 waves = 128 q rows.
// K/V 32x64 tiles double-buffered in LDS via async DMA.
// mask_mode 0: tgt_mask[Sq,Sk]; 1: src_mask[B,Sk].
// ---------------------------------------------------------------------------
__global__ __launch_bounds__(256) void flash_attn(
    const bf16* __restrict__ Q, const bf16* __restrict__ Kb,
    const bf16* __restrict__ Vb, const int* __restrict__ mask,
    int mask_mode, bf16* __restrict__ O) {
  __shared__ __align__(16) bf16 sK[2][32 * 64];
  __shared__ __align__(16) bf16 sV[2][32 * 64];
  __shared__ __align__(16) bf16 sP[8][16 * 32];

  const int wg = blockIdx.x;
  const int qb = wg & 15;
  const int bh = wg >> 4;
  const int h = bh % NHEAD, b = bh / NHEAD;
  const int wave = threadIdx.x >> 5;
  const int q0 = qb * 128 + wave * 16;
  const int lane = threadIdx.x & 31, half = lane >> 4, nl = lane & 15;
  const float scale = 0.125f;   // 1/sqrt(64)

  const bf16* qbase = Q + ((size_t)(b * SEQ + q0)) * D_MODEL + h * DKH;
  v16bf qf[2];
  qf[0] = frag_k_major(qbase, D_MODEL);
  qf[1] = frag_k_major(qbase + 32, D_MODEL);

  auto stage = [&](int t, int buf) {   // 32x64 K tile + V tile, 2 b128/thread
    int idx = threadIdx.x * 8;
    int rr = idx >> 6, cc = idx & 63;
    size_t g = ((size_t)(b * SEQ + (t << 5) + rr)) * D_MODEL + h * DKH + cc;
    g2l_b128(&sK[buf][idx], &Kb[g]);
    g2l_b128(&sV[buf][idx], &Vb[g]);
  };

  v8f acc[4] = {};
  float mrun[8], lrun[8];
#pragma unroll
  for (int r = 0; r < 8; ++r) { mrun[r] = -1e30f; lrun[r] = 0.0f; }

  const int T = SEQ / 32;
  stage(0, 0);
  for (int t = 0; t < T; ++t) {
    const int cur = t & 1;
    const int kt = t << 5;
    if (t + 1 < T) { stage(t + 1, cur ^ 1); wait_async<2>(); }
    else           { wait_async<0>(); }
    __syncthreads();

    // S tile 16x32 = Q(16x64) @ K^T
    v8f sc[2] = {};
#pragma unroll
    for (int ns = 0; ns < 2; ++ns) {
      v16bf kf0 = frag_k_major(&sK[cur][(ns * 16) * 64], 64);
      v16bf kf1 = frag_k_major(&sK[cur][(ns * 16) * 64 + 32], 64);
      sc[ns] = wmma_bf16(qf[0], kf0, sc[ns]);
      sc[ns] = wmma_bf16(qf[1], kf1, sc[ns]);
    }

    // online softmax over 32 new columns
    float p0[8], p1[8];
#pragma unroll
    for (int r = 0; r < 8; ++r) {
      int m = q0 + r + 8 * half;
      float v0 = sc[0][r] * scale, v1 = sc[1][r] * scale;
      int c0 = kt + nl, c1 = kt + 16 + nl;
      if (mask_mode == 0) {
        if (mask[(size_t)m * SEQ + c0] == 0) v0 = -1e9f;
        if (mask[(size_t)m * SEQ + c1] == 0) v1 = -1e9f;
      } else {
        if (mask[(size_t)b * SEQ + c0] == 0) v0 = -1e9f;
        if (mask[(size_t)b * SEQ + c1] == 0) v1 = -1e9f;
      }
      float mx = fmaxf(v0, v1);
      mx = fmaxf(mx, __shfl_xor(mx, 1, 32));
      mx = fmaxf(mx, __shfl_xor(mx, 2, 32));
      mx = fmaxf(mx, __shfl_xor(mx, 4, 32));
      mx = fmaxf(mx, __shfl_xor(mx, 8, 32));
      float mnew = fmaxf(mrun[r], mx);
      float e0 = __expf(v0 - mnew), e1 = __expf(v1 - mnew);
      p0[r] = e0; p1[r] = e1;
      float rs = e0 + e1;
      rs += __shfl_xor(rs, 1, 32); rs += __shfl_xor(rs, 2, 32);
      rs += __shfl_xor(rs, 4, 32); rs += __shfl_xor(rs, 8, 32);
      float alpha = __expf(mrun[r] - mnew);
      lrun[r] = lrun[r] * alpha + rs;
      mrun[r] = mnew;
#pragma unroll
      for (int j = 0; j < 4; ++j) acc[j][r] *= alpha;
    }

    // P (C-layout) -> A-fragment layout via per-wave LDS scratch
    bf16* ps = sP[wave];
#pragma unroll
    for (int r = 0; r < 8; ++r) {
      int m = r + 8 * half;
      ps[m * 32 + nl]      = f2bf(p0[r]);
      ps[m * 32 + 16 + nl] = f2bf(p1[r]);
    }
    asm volatile("s_wait_dscnt 0" ::: "memory");   // wave-local LDS RAW fence
    v16bf pf = frag_k_major(ps, 32);

    // O(16x64) += P(16x32) @ V(32x64)
#pragma unroll
    for (int j = 0; j < 4; ++j) {
      v16bf vf = frag_b_colmajor(&sV[cur][j * 16], 64);
      acc[j] = wmma_bf16(pf, vf, acc[j]);
    }
    __syncthreads();
  }

#pragma unroll
  for (int j = 0; j < 4; ++j)
#pragma unroll
    for (int r = 0; r < 8; ++r) {
      int m = q0 + r + 8 * half;
      float inv = 1.0f / lrun[r];
      size_t g = ((size_t)(b * SEQ + m)) * D_MODEL + h * DKH + j * 16 + nl;
      O[g] = f2bf(acc[j][r] * inv);
    }
}

// ---------------------------------------------------------------------------
// out = LayerNorm(X + R) * g + b ; f32 (residual stream) + optional bf16.
// ---------------------------------------------------------------------------
__global__ __launch_bounds__(256) void add_ln(
    const float* __restrict__ X, const float* __restrict__ R,
    const float* __restrict__ g, const float* __restrict__ bta,
    float* __restrict__ outF, bf16* __restrict__ outB) {
  __shared__ float rs[8], rs2[8], stat[2];
  const size_t row = (size_t)blockIdx.x * D_MODEL;
  float s = 0.0f, s2 = 0.0f;
  for (int i = threadIdx.x; i < D_MODEL; i += 256) {
    float v = X[row + i] + R[row + i];
    s += v; s2 += v * v;
  }
#pragma unroll
  for (int m = 16; m >= 1; m >>= 1) {
    s += __shfl_xor(s, m, 32); s2 += __shfl_xor(s2, m, 32);
  }
  if ((threadIdx.x & 31) == 0) { rs[threadIdx.x >> 5] = s; rs2[threadIdx.x >> 5] = s2; }
  __syncthreads();
  if (threadIdx.x == 0) {
    float a = 0.0f, b2 = 0.0f;
#pragma unroll
    for (int w = 0; w < 8; ++w) { a += rs[w]; b2 += rs2[w]; }
    float mu = a / D_MODEL;
    float var = b2 / D_MODEL - mu * mu;
    stat[0] = mu; stat[1] = rsqrtf(var + 1e-5f);
  }
  __syncthreads();
  float mu = stat[0], inv = stat[1];
  for (int i = threadIdx.x; i < D_MODEL; i += 256) {
    float v = (X[row + i] + R[row + i] - mu) * inv * g[i] + bta[i];
    if (outF) outF[row + i] = v;
    if (outB) outB[row + i] = f2bf(v);
  }
}

// ---------------------------------------------------------------------------
// Host pipeline
// ---------------------------------------------------------------------------
extern "C" void kernel_launch(void* const* d_in, const int* in_sizes, int n_in,
                              void* d_out, int out_size, void* d_ws, size_t ws_size,
                              hipStream_t stream) {
  (void)in_sizes; (void)n_in; (void)out_size; (void)ws_size;
  const float* x   = (const float*)d_in[0];
  const float* enc = (const float*)d_in[1];
  const int* tgt_mask = (const int*)d_in[2];
  const int* src_mask = (const int*)d_in[3];
  const float *sa_Wq = (const float*)d_in[4],  *sa_bq = (const float*)d_in[5];
  const float *sa_Wk = (const float*)d_in[6],  *sa_bk = (const float*)d_in[7];
  const float *sa_Wv = (const float*)d_in[8],  *sa_bv = (const float*)d_in[9];
  const float *sa_Wo = (const float*)d_in[10], *sa_bo = (const float*)d_in[11];
  const float *ca_Wq = (const float*)d_in[12], *ca_bq = (const float*)d_in[13];
  const float *ca_Wk = (const float*)d_in[14], *ca_bk = (const float*)d_in[15];
  const float *ca_Wv = (const float*)d_in[16], *ca_bv = (const float*)d_in[17];
  const float *ca_Wo = (const float*)d_in[18], *ca_bo = (const float*)d_in[19];
  const float *ff_W1 = (const float*)d_in[20], *ff_b1 = (const float*)d_in[21];
  const float *ff_W2 = (const float*)d_in[22], *ff_b2 = (const float*)d_in[23];
  const float *ln1_g = (const float*)d_in[24], *ln1_b = (const float*)d_in[25];
  const float *ln2_g = (const float*)d_in[26], *ln2_b = (const float*)d_in[27];
  const float *ln3_g = (const float*)d_in[28], *ln3_b = (const float*)d_in[29];

  char* w = (char*)d_ws;
  auto alloc = [&](size_t bytes) -> char* {
    char* p = w;
    w += (bytes + 255) & ~(size_t)255;
    return p;
  };
  const size_t TD = (size_t)NTOK * D_MODEL;
  const size_t WW = (size_t)D_MODEL * D_MODEL;
  const size_t WF = (size_t)D_FF * D_MODEL;

  bf16* xb   = (bf16*)alloc(TD * 2);
  bf16* encb = (bf16*)alloc(TD * 2);
  bf16* wsaq = (bf16*)alloc(WW * 2); bf16* wsak = (bf16*)alloc(WW * 2);
  bf16* wsav = (bf16*)alloc(WW * 2); bf16* wsao = (bf16*)alloc(WW * 2);
  bf16* wcaq = (bf16*)alloc(WW * 2); bf16* wcak = (bf16*)alloc(WW * 2);
  bf16* wcav = (bf16*)alloc(WW * 2); bf16* wcao = (bf16*)alloc(WW * 2);
  bf16* wff1 = (bf16*)alloc(WF * 2); bf16* wff2 = (bf16*)alloc(WF * 2);
  bf16* Qb   = (bf16*)alloc(TD * 2);
  bf16* Kb   = (bf16*)alloc(TD * 2);
  bf16* Vb   = (bf16*)alloc(TD * 2);
  bf16* AOb  = (bf16*)alloc(TD * 2);
  bf16* Hb   = (bf16*)alloc(TD * 2);
  bf16* FFb  = (bf16*)alloc((size_t)NTOK * D_FF * 2);
  float* T0  = (float*)alloc(TD * 4);
  float* X1  = (float*)alloc(TD * 4);
  float* X2  = (float*)alloc(TD * 4);
  float* OUT = (float*)d_out;

  auto cvt = [&](const float* src, bf16* dst, size_t n) {
    cvt_f32_bf16<<<dim3((unsigned)((n + 255) / 256)), 256, 0, stream>>>(src, dst, (int)n);
  };
  cvt(x, xb, TD); cvt(enc, encb, TD);
  cvt(sa_Wq, wsaq, WW); cvt(sa_Wk, wsak, WW); cvt(sa_Wv, wsav, WW); cvt(sa_Wo, wsao, WW);
  cvt(ca_Wq, wcaq, WW); cvt(ca_Wk, wcak, WW); cvt(ca_Wv, wcav, WW); cvt(ca_Wo, wcao, WW);
  cvt(ff_W1, wff1, WF); cvt(ff_W2, wff2, WF);

  const dim3 gProj((NTOK / 128) * (D_MODEL / 128));  // 192
  const dim3 gFF1((NTOK / 128) * (D_FF / 128));      // 768
  const dim3 gAttn(BATCH * NHEAD * (SEQ / 128));     // 384
  const dim3 gLN(NTOK);                              // 4096

  // --- self-attention ---
  gemm_bf16_wmma<<<gProj, 256, 0, stream>>>(xb, wsaq, sa_bq, nullptr, Qb,
                                            NTOK, D_MODEL, D_MODEL, 0);
  gemm_bf16_wmma<<<gProj, 256, 0, stream>>>(xb, wsak, sa_bk, nullptr, Kb,
                                            NTOK, D_MODEL, D_MODEL, 0);
  gemm_bf16_wmma<<<gProj, 256, 0, stream>>>(xb, wsav, sa_bv, nullptr, Vb,
                                            NTOK, D_MODEL, D_MODEL, 0);
  flash_attn<<<gAttn, 256, 0, stream>>>(Qb, Kb, Vb, tgt_mask, 0, AOb);
  gemm_bf16_wmma<<<gProj, 256, 0, stream>>>(AOb, wsao, sa_bo, T0, nullptr,
                                            NTOK, D_MODEL, D_MODEL, 0);
  add_ln<<<gLN, 256, 0, stream>>>(x, T0, ln1_g, ln1_b, X1, Hb);

  // --- cross-attention ---
  gemm_bf16_wmma<<<gProj, 256, 0, stream>>>(Hb, wcaq, ca_bq, nullptr, Qb,
                                            NTOK, D_MODEL, D_MODEL, 0);
  gemm_bf16_wmma<<<gProj, 256, 0, stream>>>(encb, wcak, ca_bk, nullptr, Kb,
                                            NTOK, D_MODEL, D_MODEL, 0);
  gemm_bf16_wmma<<<gProj, 256, 0, stream>>>(encb, wcav, ca_bv, nullptr, Vb,
                                            NTOK, D_MODEL, D_MODEL, 0);
  flash_attn<<<gAttn, 256, 0, stream>>>(Qb, Kb, Vb, src_mask, 1, AOb);
  gemm_bf16_wmma<<<gProj, 256, 0, stream>>>(AOb, wcao, ca_bo, T0, nullptr,
                                            NTOK, D_MODEL, D_MODEL, 0);
  add_ln<<<gLN, 256, 0, stream>>>(X1, T0, ln2_g, ln2_b, X2, Hb);

  // --- FFN ---
  gemm_bf16_wmma<<<gFF1, 256, 0, stream>>>(Hb, wff1, ff_b1, nullptr, FFb,
                                           NTOK, D_FF, D_MODEL, 1);
  gemm_bf16_wmma<<<gProj, 256, 0, stream>>>(FFb, wff2, ff_b2, T0, nullptr,
                                            NTOK, D_MODEL, D_FF, 0);
  add_ln<<<gLN, 256, 0, stream>>>(X2, T0, ln3_g, ln3_b, OUT, nullptr);
}
